// NeighborAwareModel_5729486373484
// MI455X (gfx1250) — matrix operand order repaired and checked by reading, not compiled, not run on hardware
//
#include <hip/hip_runtime.h>
#include <stdint.h>
#include <math.h>

#define B_SZ   16384
#define K_NB   5
#define S_LEN  6
#define D_DIM  128
#define DFF_D  512
#define MTOK   (B_SZ * S_LEN)   // 98304 token rows
#define QKV_N  384
#define QKV_K  128
#define QKV_HALF_N 192          // weights staged to LDS in two 48KB halves

typedef unsigned short ushort_t;
typedef __attribute__((ext_vector_type(8)))  float          v8f;
typedef __attribute__((ext_vector_type(16))) __bf16         v16bf;
typedef __attribute__((ext_vector_type(4)))  unsigned int   u32x4;
typedef __attribute__((ext_vector_type(4)))  int            i32x4;
typedef __attribute__((ext_vector_type(8)))  int            i32x8;
typedef __attribute__((ext_vector_type(4)))  float          f32x4;
typedef __attribute__((ext_vector_type(4)))  unsigned short u16x4;

union FragU { v16bf v; u32x4 q[2]; };

__device__ __forceinline__ float bf2f(unsigned int lo16) {
    return __uint_as_float(lo16 << 16);
}
__device__ __forceinline__ ushort_t f2bf(float f) {
    unsigned u = __float_as_uint(f);
    u += 0x7FFFu + ((u >> 16) & 1u);      // round-to-nearest-even
    return (ushort_t)(u >> 16);
}
__device__ __forceinline__ float wave_sum(float v) {
#pragma unroll
    for (int o = 16; o > 0; o >>= 1) v += __shfl_xor(v, o, 32);
    return v;
}

// ---------------------------------------------------------------------------
// f32 -> bf16 weight conversion
// ---------------------------------------------------------------------------
__global__ void cvt_bf16_kernel(const float* __restrict__ src,
                                ushort_t* __restrict__ dst, int n) {
    int i = blockIdx.x * 256 + threadIdx.x;
    if (i < n) dst[i] = f2bf(src[i]);
}

// ---------------------------------------------------------------------------
// Gather embedding tokens + LayerNorm1. One wave per token row (D=128).
// ---------------------------------------------------------------------------
__global__ __launch_bounds__(256) void gather_ln_kernel(
    const int* __restrict__ sel, const int* __restrict__ topk,
    const float* __restrict__ emb,
    const float* __restrict__ nw, const float* __restrict__ nb,
    int* __restrict__ ids, float* __restrict__ tok, ushort_t* __restrict__ ln)
{
    const int t    = blockIdx.x * 8 + (threadIdx.x >> 5);   // token row
    const int lane = threadIdx.x & 31;
    const int b = t / S_LEN, s = t % S_LEN;
    const int sid = sel[b];
    const int id  = (s == 0) ? sid : topk[(size_t)sid * K_NB + (s - 1)];
    if (lane == 0) ids[t] = id;

    f32x4 v = ((const f32x4*)(emb + (size_t)id * D_DIM))[lane];
    float mean = wave_sum(v.x + v.y + v.z + v.w) * (1.0f / D_DIM);
    f32x4 c = v - mean;
    float var = wave_sum(c.x*c.x + c.y*c.y + c.z*c.z + c.w*c.w) * (1.0f / D_DIM);
    float rstd = rsqrtf(var + 1e-5f);

    ((f32x4*)(tok + (size_t)t * D_DIM))[lane] = v;
    const int ci = lane * 4;
    u16x4 o;
    o.x = f2bf(c.x * rstd * nw[ci+0] + nb[ci+0]);
    o.y = f2bf(c.y * rstd * nw[ci+1] + nb[ci+1]);
    o.z = f2bf(c.z * rstd * nw[ci+2] + nb[ci+2]);
    o.w = f2bf(c.w * rstd * nw[ci+3] + nb[ci+3]);
    ((u16x4*)(ln + (size_t)t * D_DIM))[lane] = o;
}

// ---------------------------------------------------------------------------
// Generic row LayerNorm over f32 rows of width 128 -> bf16 rows.
// ---------------------------------------------------------------------------
__global__ __launch_bounds__(256) void ln_rows_kernel(
    const float* __restrict__ x, long long xstride,
    const float* __restrict__ nw, const float* __restrict__ nb,
    ushort_t* __restrict__ y, long long ystride, int rows)
{
    const int t = blockIdx.x * 8 + (threadIdx.x >> 5);
    if (t >= rows) return;
    const int lane = threadIdx.x & 31;
    f32x4 v = ((const f32x4*)(x + (size_t)t * xstride))[lane];
    float mean = wave_sum(v.x + v.y + v.z + v.w) * (1.0f / D_DIM);
    f32x4 c = v - mean;
    float var = wave_sum(c.x*c.x + c.y*c.y + c.z*c.z + c.w*c.w) * (1.0f / D_DIM);
    float rstd = rsqrtf(var + 1e-5f);
    const int ci = lane * 4;
    u16x4 o;
    o.x = f2bf(c.x * rstd * nw[ci+0] + nb[ci+0]);
    o.y = f2bf(c.y * rstd * nw[ci+1] + nb[ci+1]);
    o.z = f2bf(c.z * rstd * nw[ci+2] + nb[ci+2]);
    o.w = f2bf(c.w * rstd * nw[ci+3] + nb[ci+3]);
    ((u16x4*)(y + (size_t)t * ystride))[lane] = o;
}

// ---------------------------------------------------------------------------
// QKV GEMM with TDM weight staging:
//   out[98304,384] = A[98304,128] @ W[384,128]^T + bias
// Block = 8 M-tiles (one per wave). Each wave holds its 16x128 A tile in
// registers (read once from global) and sweeps all N-tiles; B fragments come
// from LDS. W is staged to LDS via tensor_load_to_lds in two 48KB halves.
// ---------------------------------------------------------------------------
__global__ __launch_bounds__(256) void qkv_gemm_tdm_kernel(
    const ushort_t* __restrict__ A, const ushort_t* __restrict__ W,
    const float* __restrict__ bias, ushort_t* __restrict__ out)
{
    __shared__ alignas(16) ushort_t smem[QKV_HALF_N * QKV_K];   // 48 KB

    const int lane = threadIdx.x & 31;
    const int wv   = threadIdx.x >> 5;
    const int r15  = lane & 15, hi = lane >> 4;

    // --- A tile resident in registers: 16 rows x 128 K (32 VGPRs/lane) ---
    const int mt = blockIdx.x * 8 + wv;             // grid = MTOK/128 (exact)
    const ushort_t* arow = A + (size_t)(mt * 16 + r15) * QKV_K;
    FragU a[4];
#pragma unroll
    for (int ks = 0; ks < 4; ks++) {
        a[ks].q[0] = *(const u32x4*)(arow + ks * 32 + hi * 16);
        a[ks].q[1] = *(const u32x4*)(arow + ks * 32 + hi * 16 + 8);
    }

    const size_t obase = (size_t)mt * 16 * QKV_N;

    for (int half = 0; half < 2; half++) {
        // ---- Tensor Data Mover: DMA 192x128 bf16 weight slab into LDS ----
        if (wv == 0) {
            const uint64_t ga = (uint64_t)(uintptr_t)(W + (size_t)half * QKV_HALF_N * QKV_K);
            u32x4 g0;
            g0[0] = 1u;                                   // count=1, user mode, no gather
            g0[1] = 0u;                                   // lds_addr (sole LDS object => 0)
            g0[2] = (uint32_t)ga;                         // global_addr[31:0]
            g0[3] = (uint32_t)((ga >> 32) & 0x01FFFFFFu)  // global_addr[56:32]
                  | (2u << 30);                           // type = 2 ("image")
            i32x8 g1;
            g1[0] = 0x00010000;                           // data_size = 2B
            g1[1] = (int)(QKV_K << 16);                   // tensor_dim0 = 128
            g1[2] = (int)(QKV_HALF_N << 16);              // tensor_dim1 = 192
            g1[3] = (int)(QKV_K << 16);                   // tile_dim0 = 128
            g1[4] = (int)QKV_HALF_N;                      // tile_dim1 = 192, tile_dim2 = 0
            g1[5] = (int)QKV_K;                           // tensor_dim0_stride = 128
            g1[6] = 0;
            g1[7] = 0;
            i32x4 gz = {0, 0, 0, 0};
#if defined(__clang_major__) && (__clang_major__ >= 23)
            i32x8 gz8 = {0, 0, 0, 0, 0, 0, 0, 0};
            __builtin_amdgcn_tensor_load_to_lds(g0, g1, gz, gz, gz8, 0);
#else
            __builtin_amdgcn_tensor_load_to_lds(g0, g1, gz, gz, 0);
#endif
            __builtin_amdgcn_s_wait_tensorcnt((short)0);
        }
        __syncthreads();
        // The TDM write to LDS is invisible to LLVM: the LDS address travels
        // only as opaque descriptor *data*, so `smem` looks like a never-
        // stored internal global and its loads fold to undef. Escape the
        // pointer INTO the asm so GlobalOpt must assume it may be written.
        asm volatile("" :: "v"(&smem[0]) : "memory");

        // ---- sweep 12 N-tiles of this half, 2 accumulator chains at a time ----
        for (int nt2 = 0; nt2 < QKV_HALF_N / 32; nt2++) {
            const int nloc = nt2 * 32;                    // column within LDS slab
            const int n0   = half * QKV_HALF_N + nloc;    // global output column
            const ushort_t* w0 = smem + (size_t)(nloc + r15) * QKV_K;
            const ushort_t* w1 = w0 + 16 * QKV_K;
            v8f acc0 = {}; v8f acc1 = {};
#pragma unroll
            for (int ks = 0; ks < 4; ks++) {
                FragU b0, b1;
                b0.q[0] = *(const u32x4*)(w0 + ks * 32 + hi * 16);
                b0.q[1] = *(const u32x4*)(w0 + ks * 32 + hi * 16 + 8);
                b1.q[0] = *(const u32x4*)(w1 + ks * 32 + hi * 16);
                b1.q[1] = *(const u32x4*)(w1 + ks * 32 + hi * 16 + 8);
                acc0 = __builtin_amdgcn_wmma_f32_16x16x32_bf16(false, a[ks].v, false, b0.v,
                                                               (short)0, acc0, false, false);
                acc1 = __builtin_amdgcn_wmma_f32_16x16x32_bf16(false, a[ks].v, false, b1.v,
                                                               (short)0, acc1, false, false);
            }
            const float bv0 = bias[n0 + r15];
            const float bv1 = bias[n0 + 16 + r15];
#pragma unroll
            for (int r = 0; r < 8; r++) {
                const size_t rowb = obase + (size_t)(8 * hi + r) * QKV_N;
                out[rowb + n0 + r15]      = f2bf(acc0[r] + bv0);
                out[rowb + n0 + 16 + r15] = f2bf(acc1[r] + bv1);
            }
        }
        asm volatile("" :: "v"(&smem[0]) : "memory");
        __syncthreads();   // all waves done reading before next TDM overwrite
    }
}

// ---------------------------------------------------------------------------
// Generic WMMA bf16 GEMM: out[M,N] = act( A[M,K] @ W[N,K]^T + bias (+resid) )
// One wave per 16x32 output tile, fragments loaded directly from global.
// ---------------------------------------------------------------------------
enum { EPI_LIN = 0, EPI_GELU = 1, EPI_RELU = 2, EPI_RES = 3 };

template<int KDIM, int EPI>
__global__ __launch_bounds__(256) void gemm_bf16_wmma(
    const ushort_t* __restrict__ A, const ushort_t* __restrict__ W,
    const float* __restrict__ bias, const float* __restrict__ resid,
    int residStride, void* __restrict__ outp, int M, int N)
{
    const int lane = threadIdx.x & 31;
    const int wv   = threadIdx.x >> 5;
    const long long tile = (long long)blockIdx.x * 8 + wv;
    const int npairs = N >> 5;
    const long long mt = tile / npairs;
    const int np = (int)(tile - mt * npairs);
    if (mt >= (long long)(M >> 4)) return;          // whole wave exits together
    const int n0 = np * 32;
    const long long m0 = mt * 16;

    const int r15 = lane & 15, hi = lane >> 4;
    const ushort_t* arow = A + (size_t)(m0 + r15) * KDIM;
    const ushort_t* w0   = W + (size_t)(n0 + r15) * KDIM;
    const ushort_t* w1   = w0 + (size_t)16 * KDIM;

    v8f acc0 = {}; v8f acc1 = {};
#pragma unroll
    for (int k0 = 0; k0 < KDIM; k0 += 32) {
        FragU a, b0, b1;
        // A 16x32 bf16: lanes 0-15 hold K {0..15}, lanes 16-31 {16..31}
        a.q[0]  = *(const u32x4*)(arow + k0 + hi * 16);
        a.q[1]  = *(const u32x4*)(arow + k0 + hi * 16 + 8);
        // B 32x16 bf16: lanes 0-15 hold K {0..15}, lanes 16-31 {16..31}
        b0.q[0] = *(const u32x4*)(w0 + k0 + hi * 16);
        b0.q[1] = *(const u32x4*)(w0 + k0 + hi * 16 + 8);
        b1.q[0] = *(const u32x4*)(w1 + k0 + hi * 16);
        b1.q[1] = *(const u32x4*)(w1 + k0 + hi * 16 + 8);
        acc0 = __builtin_amdgcn_wmma_f32_16x16x32_bf16(false, a.v, false, b0.v,
                                                       (short)0, acc0, false, false);
        acc1 = __builtin_amdgcn_wmma_f32_16x16x32_bf16(false, a.v, false, b1.v,
                                                       (short)0, acc1, false, false);
    }

    const float bv0 = bias[n0 + r15];
    const float bv1 = bias[n0 + 16 + r15];
#pragma unroll
    for (int r = 0; r < 8; r++) {
        const long long row = m0 + 8 * hi + r;
        const size_t base = (size_t)row * N;
        float v0 = acc0[r] + bv0;
        float v1 = acc1[r] + bv1;
        if (EPI == EPI_RES) {
            const size_t rb = (size_t)row * residStride;
            v0 += resid[rb + n0 + r15];
            v1 += resid[rb + n0 + 16 + r15];
        }
        if (EPI == EPI_GELU) {
            v0 = 0.5f * v0 * (1.0f + erff(v0 * 0.70710678f));
            v1 = 0.5f * v1 * (1.0f + erff(v1 * 0.70710678f));
        }
        if (EPI == EPI_RELU) { v0 = fmaxf(v0, 0.0f); v1 = fmaxf(v1, 0.0f); }
        if (EPI == EPI_RES) {
            ((float*)outp)[base + n0 + r15]      = v0;
            ((float*)outp)[base + n0 + 16 + r15] = v1;
        } else {
            ((ushort_t*)outp)[base + n0 + r15]      = f2bf(v0);
            ((ushort_t*)outp)[base + n0 + 16 + r15] = f2bf(v1);
        }
    }
}

// ---------------------------------------------------------------------------
// Attention for query token 0 only (only token 0 is consumed downstream).
// One thread per (batch, head): 6 masked dot-64s, softmax, weighted V sum.
// ---------------------------------------------------------------------------
__global__ __launch_bounds__(256) void attn_q0_kernel(
    const ushort_t* __restrict__ qkv, const int* __restrict__ ids,
    ushort_t* __restrict__ out)  // out: [B,128] bf16
{
    const int tid = blockIdx.x * 256 + threadIdx.x;   // B*2 threads exactly
    const int h = tid & 1;
    const int b = tid >> 1;
    const ushort_t* seq = qkv + (size_t)b * S_LEN * QKV_N;

    float q[64];
    {
        const u32x4* qp = (const u32x4*)(seq + h * 64);   // token 0
#pragma unroll
        for (int i = 0; i < 8; i++) {
            u32x4 t = qp[i];
#pragma unroll
            for (int c = 0; c < 4; c++) {
                unsigned w = t[c];
                q[i * 8 + 2 * c]     = bf2f(w & 0xFFFFu);
                q[i * 8 + 2 * c + 1] = bf2f(w >> 16);
            }
        }
    }

    float sc[S_LEN];
#pragma unroll
    for (int j = 0; j < S_LEN; j++) {
        if (ids[b * S_LEN + j] == 0) { sc[j] = -1e9f; continue; }
        const u32x4* kp = (const u32x4*)(seq + j * QKV_N + 128 + h * 64);
        float acc = 0.0f;
#pragma unroll
        for (int i = 0; i < 8; i++) {
            u32x4 t = kp[i];
#pragma unroll
            for (int c = 0; c < 4; c++) {
                unsigned w = t[c];
                acc += q[i * 8 + 2 * c]     * bf2f(w & 0xFFFFu);
                acc += q[i * 8 + 2 * c + 1] * bf2f(w >> 16);
            }
        }
        sc[j] = acc * 0.125f;   // 1/sqrt(64)
    }

    float m = sc[0];
#pragma unroll
    for (int j = 1; j < S_LEN; j++) m = fmaxf(m, sc[j]);
    float p[S_LEN], denom = 0.0f;
#pragma unroll
    for (int j = 0; j < S_LEN; j++) { p[j] = expf(sc[j] - m); denom += p[j]; }
    const float inv = 1.0f / denom;
#pragma unroll
    for (int j = 0; j < S_LEN; j++) p[j] *= inv;

    float o[64];
#pragma unroll
    for (int d = 0; d < 64; d++) o[d] = 0.0f;
#pragma unroll
    for (int j = 0; j < S_LEN; j++) {
        const u32x4* vp = (const u32x4*)(seq + j * QKV_N + 256 + h * 64);
        const float pj = p[j];
#pragma unroll
        for (int i = 0; i < 8; i++) {
            u32x4 t = vp[i];
#pragma unroll
            for (int c = 0; c < 4; c++) {
                unsigned w = t[c];
                o[i * 8 + 2 * c]     += pj * bf2f(w & 0xFFFFu);
                o[i * 8 + 2 * c + 1] += pj * bf2f(w >> 16);
            }
        }
    }

    u32x4* op = (u32x4*)(out + (size_t)b * D_DIM + h * 64);
#pragma unroll
    for (int i = 0; i < 8; i++) {
        u32x4 pk;
#pragma unroll
        for (int c = 0; c < 4; c++) {
            pk[c] = (unsigned)f2bf(o[i * 8 + 2 * c])
                  | ((unsigned)f2bf(o[i * 8 + 2 * c + 1]) << 16);
        }
        op[i] = pk;
    }
}

// ---------------------------------------------------------------------------
// Final MLP layer: out[b] = h2[b,:] . w3 + b3   (64-wide dot per thread)
// ---------------------------------------------------------------------------
__global__ void final_dot_kernel(const ushort_t* __restrict__ h2,
                                 const float* __restrict__ w3,
                                 const float* __restrict__ b3,
                                 float* __restrict__ out)
{
    const int b = blockIdx.x * 256 + threadIdx.x;
    float acc = b3[0];
    const ushort_t* row = h2 + (size_t)b * 64;
#pragma unroll
    for (int d = 0; d < 64; d++) acc += bf2f((unsigned)row[d]) * w3[d];
    out[b] = acc;
}

// ---------------------------------------------------------------------------
// Host launcher
// ---------------------------------------------------------------------------
extern "C" void kernel_launch(void* const* d_in, const int* in_sizes, int n_in,
                              void* d_out, int out_size, void* d_ws, size_t ws_size,
                              hipStream_t stream)
{
    (void)in_sizes; (void)n_in; (void)out_size; (void)ws_size;

    const int*   user_id   = (const int*)  d_in[0];
    const int*   item_id   = (const int*)  d_in[1];
    const int*   user_topk = (const int*)  d_in[2];
    const int*   item_topk = (const int*)  d_in[3];
    const float* user_emb  = (const float*)d_in[4];
    const float* item_emb  = (const float*)d_in[5];
    // u_agg @6..19, i_agg @20..33 (dict order: in_w,in_b,out_w,out_b,l1_w,l1_b,
    // l2_w,l2_b,n1_w,n1_b,n2_w,n2_b,on_w,on_b); mlp @34..39 (w1,b1,w2,b2,w3,b3)
    const float* mlp_w1 = (const float*)d_in[34];
    const float* mlp_b1 = (const float*)d_in[35];
    const float* mlp_w2 = (const float*)d_in[36];
    const float* mlp_b2 = (const float*)d_in[37];
    const float* mlp_w3 = (const float*)d_in[38];
    const float* mlp_b3 = (const float*)d_in[39];

    char* ws = (char*)d_ws;
    size_t off = 0;
    auto alloc = [&](size_t bytes) -> char* {
        char* p = ws + off;
        off += (bytes + 255) & ~(size_t)255;
        return p;
    };

    // bf16 weight copies
    ushort_t* w_in_bf[2];  ushort_t* w_out_bf[2];
    ushort_t* w_l1_bf[2];  ushort_t* w_l2_bf[2];
    for (int s = 0; s < 2; s++) {
        w_in_bf[s]  = (ushort_t*)alloc(384 * 128 * 2);
        w_out_bf[s] = (ushort_t*)alloc(128 * 128 * 2);
        w_l1_bf[s]  = (ushort_t*)alloc(512 * 128 * 2);
        w_l2_bf[s]  = (ushort_t*)alloc(128 * 512 * 2);
    }
    ushort_t* w1_bf = (ushort_t*)alloc(128 * 256 * 2);
    ushort_t* w2_bf = (ushort_t*)alloc(64 * 128 * 2);

    // activations (reused across user/item sides)
    int*      ids_buf = (int*)     alloc((size_t)MTOK * 4);
    float*    tok_buf = (float*)   alloc((size_t)MTOK * D_DIM * 4);
    ushort_t* ln_buf  = (ushort_t*)alloc((size_t)MTOK * D_DIM * 2);
    ushort_t* qkv_buf = (ushort_t*)alloc((size_t)MTOK * QKV_N * 2);
    ushort_t* att_buf = (ushort_t*)alloc((size_t)B_SZ * D_DIM * 2);
    float*    h_buf   = (float*)   alloc((size_t)B_SZ * D_DIM * 4);
    ushort_t* ff1_buf = (ushort_t*)alloc((size_t)B_SZ * DFF_D * 2);
    ushort_t* ln2_buf = (ushort_t*)alloc((size_t)B_SZ * D_DIM * 2);
    ushort_t* x_mlp   = (ushort_t*)alloc((size_t)B_SZ * 256 * 2);
    ushort_t* h1_mlp  = (ushort_t*)alloc((size_t)B_SZ * 128 * 2);
    ushort_t* h2_mlp  = (ushort_t*)alloc((size_t)B_SZ * 64 * 2);

    auto cvt = [&](const float* src, ushort_t* dst, int n) {
        cvt_bf16_kernel<<<(n + 255) / 256, 256, 0, stream>>>(src, dst, n);
    };
    for (int s = 0; s < 2; s++) {
        const int base = 6 + s * 14;
        cvt((const float*)d_in[base + 0], w_in_bf[s],  384 * 128);
        cvt((const float*)d_in[base + 2], w_out_bf[s], 128 * 128);
        cvt((const float*)d_in[base + 4], w_l1_bf[s],  512 * 128);
        cvt((const float*)d_in[base + 6], w_l2_bf[s],  128 * 512);
    }
    cvt(mlp_w1, w1_bf, 128 * 256);
    cvt(mlp_w2, w2_bf, 64 * 128);

    auto gblocks = [](size_t M, size_t N) -> unsigned {
        return (unsigned)(((M >> 4) * (N >> 5) + 7) / 8);
    };

    for (int s = 0; s < 2; s++) {
        const int*   sel  = s ? item_id   : user_id;
        const int*   tpk  = s ? item_topk : user_topk;
        const float* emb  = s ? item_emb  : user_emb;
        const int base = 6 + s * 14;
        const float* in_b  = (const float*)d_in[base + 1];
        const float* out_b = (const float*)d_in[base + 3];
        const float* l1_b  = (const float*)d_in[base + 5];
        const float* l2_b  = (const float*)d_in[base + 7];
        const float* n1_w  = (const float*)d_in[base + 8];
        const float* n1_b  = (const float*)d_in[base + 9];
        const float* n2_w  = (const float*)d_in[base + 10];
        const float* n2_b  = (const float*)d_in[base + 11];
        const float* on_w  = (const float*)d_in[base + 12];
        const float* on_b  = (const float*)d_in[base + 13];

        // gather + LN1 over all B*6 tokens
        gather_ln_kernel<<<MTOK / 8, 256, 0, stream>>>(
            sel, tpk, emb, n1_w, n1_b, ids_buf, tok_buf, ln_buf);

        // QKV via TDM-staged weights: [98304,128] x [128,384]^T
        qkv_gemm_tdm_kernel<<<MTOK / 128, 256, 0, stream>>>(
            ln_buf, w_in_bf[s], in_b, qkv_buf);

        // attention, query token 0 only: B*H threads
        attn_q0_kernel<<<(B_SZ * 2) / 256, 256, 0, stream>>>(
            qkv_buf, ids_buf, att_buf);

        // out-proj + residual (token-0 rows of tok_buf, stride 6*128)
        gemm_bf16_wmma<128, EPI_RES><<<gblocks(B_SZ, 128), 256, 0, stream>>>(
            att_buf, w_out_bf[s], out_b, tok_buf, S_LEN * D_DIM, h_buf, B_SZ, 128);

        // LN2
        ln_rows_kernel<<<B_SZ / 8, 256, 0, stream>>>(
            h_buf, D_DIM, n2_w, n2_b, ln2_buf, D_DIM, B_SZ);

        // FF1 + exact GELU: [16384,128] x [128,512]^T
        gemm_bf16_wmma<128, EPI_GELU><<<gblocks(B_SZ, DFF_D), 256, 0, stream>>>(
            ln2_buf, w_l1_bf[s], l1_b, nullptr, 0, ff1_buf, B_SZ, DFF_D);

        // FF2 + residual, in place into h_buf: [16384,512] x [512,128]^T
        gemm_bf16_wmma<512, EPI_RES><<<gblocks(B_SZ, 128), 256, 0, stream>>>(
            ff1_buf, w_l2_bf[s], l2_b, h_buf, D_DIM, h_buf, B_SZ, 128);

        // output norm -> concat buffer columns [s*128, s*128+128)
        ln_rows_kernel<<<B_SZ / 8, 256, 0, stream>>>(
            h_buf, D_DIM, on_w, on_b, x_mlp + s * 128, 256, B_SZ);
    }

    // MLP: [16384,256]x[256,128]^T ReLU -> [16384,128]x[128,64]^T ReLU -> dot64
    gemm_bf16_wmma<256, EPI_RELU><<<gblocks(B_SZ, 128), 256, 0, stream>>>(
        x_mlp, w1_bf, mlp_b1, nullptr, 0, h1_mlp, B_SZ, 128);
    gemm_bf16_wmma<128, EPI_RELU><<<gblocks(B_SZ, 64), 256, 0, stream>>>(
        h1_mlp, w2_bf, mlp_b2, nullptr, 0, h2_mlp, B_SZ, 64);
    final_dot_kernel<<<B_SZ / 256, 256, 0, stream>>>(
        h2_mlp, mlp_w3, mlp_b3, (float*)d_out);
}
